// LinearAttention_9131100471582
// MI455X (gfx1250) — matrix operand order repaired and checked by reading, not compiled, not run on hardware
//
#include <hip/hip_runtime.h>
#include <cstdint>

// ---------------------------------------------------------------------------
// Linear attention for MI455X (gfx1250, wave32, WMMA + TDM).
// B=4, S=4096, D=1024, H=16, dk=64.
// Projections: TDM (tensor_load_to_lds) -> LDS double-buffer -> ds_load -> WMMA.
// Per-head KV / Q@KV GEMMs: direct global b128 loads -> WMMA.
// ---------------------------------------------------------------------------

typedef __attribute__((ext_vector_type(16))) __bf16 v16bf;
typedef __attribute__((ext_vector_type(8)))  float  v8f;
typedef __attribute__((ext_vector_type(4)))  unsigned int v4u;
typedef __attribute__((ext_vector_type(8)))  int v8i;
typedef __attribute__((ext_vector_type(4)))  int v4i;

union Frag {
  v16bf v;
  uint4 q[2];
};

__device__ __forceinline__ unsigned short f2bf(float f) {
  unsigned u = __builtin_bit_cast(unsigned, f);
  u += 0x7FFFu + ((u >> 16) & 1u);   // round-to-nearest-even
  return (unsigned short)(u >> 16);
}
__device__ __forceinline__ float bflo(unsigned u) { return __builtin_bit_cast(float, u << 16); }
__device__ __forceinline__ float bfhi(unsigned u) { return __builtin_bit_cast(float, u & 0xFFFF0000u); }

// A-operand (16x32 bf16): lane row = lane%16; elems 0..7 -> K=lhi*8+.., 8..15 -> +16.
__device__ __forceinline__ void load_a(Frag& f, const unsigned short* row, int k0, int lhi) {
  const unsigned short* p = row + k0 + lhi * 8;
  f.q[0] = *reinterpret_cast<const uint4*>(p);
  f.q[1] = *reinterpret_cast<const uint4*>(p + 16);
}
// B-operand (32x16 bf16): lane col = lane%16; elems 0..7 -> K=lhi*8+.., 8..15 -> +16.
__device__ __forceinline__ void load_b(Frag& f, const unsigned short* row, int k0, int lhi) {
  const unsigned short* p = row + k0 + lhi * 8;
  f.q[0] = *reinterpret_cast<const uint4*>(p);
  f.q[1] = *reinterpret_cast<const uint4*>(p + 16);
}

__device__ __forceinline__ v8f wmma_bf(const Frag& a, const Frag& b, v8f c) {
  return __builtin_amdgcn_wmma_f32_16x16x32_bf16(false, a.v, false, b.v, (short)0, c,
                                                 false, false);
}

// ---------------------------------------------------------------------------
// TDM: 2D tile load, bf16 elements, LDS row padding 128B data + 16B pad
// (pad_interval code 4 = 32 DWORDs, pad_amount code 3 = 4 DWORDs) -> pitch 144B.
// D# layout per CDNA5 ISA section 8 (groups 0/1; groups 2/3 zero for 2D).
// ---------------------------------------------------------------------------
__device__ __forceinline__ void tdm_load_2d(unsigned lds_off, const void* gaddr,
                                            unsigned tensor_d0, unsigned tensor_d1,
                                            unsigned tile_d0, unsigned tile_d1,
                                            unsigned stride0) {
  unsigned long long ga = (unsigned long long)(size_t)gaddr;
  v4u g0;
  g0.x = 1u;                                            // count=1 (valid), user mode
  g0.y = lds_off;                                       // lds_addr (bytes)
  g0.z = (unsigned)(ga & 0xFFFFFFFFull);                // global_addr[31:0]
  g0.w = (unsigned)((ga >> 32) & 0x01FFFFFFull) | (2u << 30);  // addr[56:32] | type=2
  v8i g1;
  g1[0] = (int)((1u << 16) | (1u << 20) | (4u << 22) | (3u << 25)); // 2B, pad 32DW+4DW
  g1[1] = (int)((tensor_d0 & 0xFFFFu) << 16);
  g1[2] = (int)((tensor_d0 >> 16) | ((tensor_d1 & 0xFFFFu) << 16));
  g1[3] = (int)((tensor_d1 >> 16) | (tile_d0 << 16));
  g1[4] = (int)(tile_d1 & 0xFFFFu);                     // tile_dim1 (tile_dim2 = 0)
  g1[5] = (int)stride0;                                 // tensor_dim0_stride[31:0]
  g1[6] = 0;
  g1[7] = 0;
  v4i z4 = {0, 0, 0, 0};
#if __clang_major__ >= 23
  v8i z8 = {0, 0, 0, 0, 0, 0, 0, 0};
  __builtin_amdgcn_tensor_load_to_lds(g0, g1, z4, z4, z8, 0);
#else
  __builtin_amdgcn_tensor_load_to_lds(g0, g1, z4, z4, 0);
#endif
}

// LDS fragment reads (pitch 144B rows; starts 36*r mod 64 banks -> conflict-free)
__device__ __forceinline__ void lds_load_a(Frag& f, const char* base, int row, int kl, int lhi) {
  const char* p = base + row * 144 + (kl + lhi * 8) * 2;
  f.q[0] = *reinterpret_cast<const uint4*>(p);
  f.q[1] = *reinterpret_cast<const uint4*>(p + 32);
}
__device__ __forceinline__ void lds_load_b(Frag& f, const char* base, int row, int kl, int lhi) {
  const char* p = base + row * 144 + (kl + lhi * 8) * 2;
  f.q[0] = *reinterpret_cast<const uint4*>(p);
  f.q[1] = *reinterpret_cast<const uint4*>(p + 32);
}

// ---------------------------------------------------------------------------
// fp32 -> bf16 conversion
// ---------------------------------------------------------------------------
__global__ __launch_bounds__(256) void cvt_bf16(const float* __restrict__ in,
                                                unsigned short* __restrict__ out, int n) {
  int i = (blockIdx.x * 256 + threadIdx.x) * 4;
  if (i >= n) return;
  float4 f = *reinterpret_cast<const float4*>(in + i);
  unsigned u0 = (unsigned)f2bf(f.x) | ((unsigned)f2bf(f.y) << 16);
  unsigned u1 = (unsigned)f2bf(f.z) | ((unsigned)f2bf(f.w) << 16);
  *reinterpret_cast<uint2*>(out + i) = make_uint2(u0, u1);
}

// ---------------------------------------------------------------------------
// Projection GEMM with TDM->LDS pipeline: C[m,n] = sum_k A[m,k]*W[n,k] + bias[n]
// MODE 0: V -> channel-major bf16; 1: Q -> row-major bf16 (elu+1);
// MODE 2: K -> channel-major bf16 (elu+1); 3: row-major fp32 (final projection).
// WG tile 128(M)x64(N); k-chunk 64; 8 waves (4x2), wave tile 32x32.
// LDS: A 128x144B, B 64x144B, double buffered = 55296 bytes (dynamic).
// ---------------------------------------------------------------------------
template <int MODE>
__global__ __launch_bounds__(256) void proj_gemm(const unsigned short* __restrict__ A,
                                                 const unsigned short* __restrict__ W,
                                                 const float* __restrict__ bias,
                                                 unsigned short* __restrict__ obf,
                                                 float* __restrict__ of32) {
  extern __shared__ char smem[];
  const int lane = threadIdx.x & 31;
  const int wv   = threadIdx.x >> 5;
  const int lrow = lane & 15;
  const int lhi  = lane >> 4;
  const int wm   = wv & 3;        // 4 waves over M
  const int wn   = wv >> 2;       // 2 waves over N
  const int bm   = blockIdx.x * 128;
  const int bn   = blockIdx.y * 64;

  const unsigned PA = 128u * 144u;            // 18432 bytes per A buffer
  const unsigned PB = 64u * 144u;             // 9216 bytes per B buffer
  const unsigned offA[2] = {0u, PA};
  const unsigned offB[2] = {2u * PA, 2u * PA + PB};

  v8f acc[2][2] = {};

  if (wv == 0) {
    tdm_load_2d(offA[0], A + (size_t)bm * 1024, 1024u, 16384u, 64u, 128u, 1024u);
    tdm_load_2d(offB[0], W + (size_t)bn * 1024, 1024u, 1024u, 64u, 64u, 1024u);
  }

  for (int kc = 0; kc < 1024; kc += 64) {
    const int cur = (kc >> 6) & 1;
    if (wv == 0) {
      if (kc + 64 < 1024) {
        tdm_load_2d(offA[cur ^ 1], A + (size_t)bm * 1024 + kc + 64,
                    1024u, 16384u, 64u, 128u, 1024u);
        tdm_load_2d(offB[cur ^ 1], W + (size_t)bn * 1024 + kc + 64,
                    1024u, 1024u, 64u, 64u, 1024u);
        __builtin_amdgcn_s_wait_tensorcnt(2);   // current buffer's two loads done
      } else {
        __builtin_amdgcn_s_wait_tensorcnt(0);
      }
    }
    __syncthreads();

    const char* pA = smem + offA[cur];
    const char* pB = smem + offB[cur];
#pragma unroll
    for (int kl = 0; kl < 64; kl += 32) {
      Frag af[2], bfr[2];
#pragma unroll
      for (int mi = 0; mi < 2; ++mi)
        lds_load_a(af[mi], pA, wm * 32 + mi * 16 + lrow, kl, lhi);
#pragma unroll
      for (int ni = 0; ni < 2; ++ni)
        lds_load_b(bfr[ni], pB, wn * 32 + ni * 16 + lrow, kl, lhi);
#pragma unroll
      for (int mi = 0; mi < 2; ++mi)
#pragma unroll
        for (int ni = 0; ni < 2; ++ni)
          acc[mi][ni] = wmma_bf(af[mi], bfr[ni], acc[mi][ni]);
    }
    __syncthreads();
  }

#pragma unroll
  for (int ni = 0; ni < 2; ++ni) {
    const int n = bn + wn * 32 + ni * 16 + lrow;
    const float bv = bias[n];
#pragma unroll
    for (int mi = 0; mi < 2; ++mi) {
#pragma unroll
      for (int r = 0; r < 8; ++r) {
        const int m = bm + wm * 32 + mi * 16 + r + 8 * lhi;
        float v = acc[mi][ni][r] + bv;
        if (MODE == 1 || MODE == 2) v = (v > 0.0f) ? (v + 1.0f) : __expf(v);  // elu(x)+1
        if (MODE == 3) {
          of32[(size_t)m * 1024 + n] = v;
        } else if (MODE == 1) {
          obf[(size_t)m * 1024 + n] = f2bf(v);
        } else {  // channel-major: [b*1024 + n][s]
          const int b = m >> 12, s = m & 4095;
          obf[((size_t)b * 1024 + n) * 4096 + s] = f2bf(v);
        }
      }
    }
  }
}

// ---------------------------------------------------------------------------
// KV state: KVt[bh][e][d] = sum_s V[s,e]*K[s,d], per-head 64x64x4096 GEMM.
// ---------------------------------------------------------------------------
__global__ __launch_bounds__(256) void kv_gemm(const unsigned short* __restrict__ Vt,
                                               const unsigned short* __restrict__ Kt,
                                               unsigned short* __restrict__ KVt) {
  const int lane = threadIdx.x & 31;
  const int wv   = threadIdx.x >> 5;
  const int lrow = lane & 15, lhi = lane >> 4;
  const int bh  = blockIdx.x;          // b*16 + h
  const int tm  = wv & 3;              // M tile (e)
  const int tn0 = (wv >> 2) * 2;       // first of 2 N tiles (d)

  v8f acc[2] = {};
  const unsigned short* arow  = Vt + (size_t)(bh * 64 + tm * 16 + lrow) * 4096;
  const unsigned short* brow0 = Kt + (size_t)(bh * 64 + tn0 * 16 + lrow) * 4096;
  const unsigned short* brow1 = brow0 + (size_t)16 * 4096;

  for (int s0 = 0; s0 < 4096; s0 += 32) {
    Frag a, b0, b1;
    load_a(a, arow, s0, lhi);
    load_b(b0, brow0, s0, lhi);
    load_b(b1, brow1, s0, lhi);
    acc[0] = wmma_bf(a, b0, acc[0]);
    acc[1] = wmma_bf(a, b1, acc[1]);
  }

#pragma unroll
  for (int j = 0; j < 2; ++j) {
    const int n = (tn0 + j) * 16 + lrow;
#pragma unroll
    for (int r = 0; r < 8; ++r) {
      const int mm = tm * 16 + r + 8 * lhi;
      KVt[(size_t)bh * 4096 + mm * 64 + n] = f2bf(acc[j][r]);
    }
  }
}

// ---------------------------------------------------------------------------
// K row sums over s: Ksum[b*1024 + c] = sum_s Kt[row][s]
// ---------------------------------------------------------------------------
__global__ __launch_bounds__(256) void ksum_kernel(const unsigned short* __restrict__ Kt,
                                                   float* __restrict__ Ksum) {
  const int row = blockIdx.x * 256 + threadIdx.x;
  const unsigned short* p = Kt + (size_t)row * 4096;
  float acc = 0.0f;
  for (int s = 0; s < 4096; s += 8) {
    uint4 qd = *reinterpret_cast<const uint4*>(p + s);
    acc += bflo(qd.x) + bfhi(qd.x) + bflo(qd.y) + bfhi(qd.y) +
           bflo(qd.z) + bfhi(qd.z) + bflo(qd.w) + bfhi(qd.w);
  }
  Ksum[row] = acc;
}

// ---------------------------------------------------------------------------
// Z[bh][s] = 1/(Q[b,s,h*64:..] . Ksum[bh*64:..] + eps)
// ---------------------------------------------------------------------------
__global__ __launch_bounds__(256) void z_kernel(const unsigned short* __restrict__ Qb,
                                                const float* __restrict__ Ksum,
                                                float* __restrict__ Z) {
  const int idx = blockIdx.x * 256 + threadIdx.x;
  const int bh = idx >> 12, s = idx & 4095;
  const int b = bh >> 4, h = bh & 15;
  const unsigned short* q = Qb + (size_t)(b * 4096 + s) * 1024 + h * 64;
  const float* ks = Ksum + bh * 64;
  float acc = 0.0f;
#pragma unroll
  for (int d = 0; d < 64; d += 8) {
    uint4 qd = *reinterpret_cast<const uint4*>(q + d);
    acc += bflo(qd.x) * ks[d + 0] + bfhi(qd.x) * ks[d + 1];
    acc += bflo(qd.y) * ks[d + 2] + bfhi(qd.y) * ks[d + 3];
    acc += bflo(qd.z) * ks[d + 4] + bfhi(qd.z) * ks[d + 5];
    acc += bflo(qd.w) * ks[d + 6] + bfhi(qd.w) * ks[d + 7];
  }
  Z[idx] = 1.0f / (acc + 1e-6f);
}

// ---------------------------------------------------------------------------
// out2[b,s,h*64+e] = Z[bh][s] * sum_d Q[b,s,h*64+d] * KVt[bh][e][d]
// ---------------------------------------------------------------------------
__global__ __launch_bounds__(256) void qkv_gemm(const unsigned short* __restrict__ Qb,
                                                const unsigned short* __restrict__ KVt,
                                                const float* __restrict__ Z,
                                                unsigned short* __restrict__ O2) {
  const int lane = threadIdx.x & 31;
  const int wv   = threadIdx.x >> 5;
  const int lrow = lane & 15, lhi = lane >> 4;
  const int bh = blockIdx.y;
  const int b = bh >> 4, h = bh & 15;
  const int m0 = blockIdx.x * 128 + wv * 16;

  v8f acc[4] = {};
  const unsigned short* arow = Qb + (size_t)(b * 4096 + m0 + lrow) * 1024 + h * 64;
#pragma unroll
  for (int k0 = 0; k0 < 64; k0 += 32) {
    Frag a;
    load_a(a, arow, k0, lhi);
#pragma unroll
    for (int ni = 0; ni < 4; ++ni) {
      Frag bb;
      load_b(bb, KVt + (size_t)bh * 4096 + (ni * 16 + lrow) * 64, k0, lhi);
      acc[ni] = wmma_bf(a, bb, acc[ni]);
    }
  }

#pragma unroll
  for (int r = 0; r < 8; ++r) {
    const int s = m0 + r + 8 * lhi;
    const float z = Z[(size_t)bh * 4096 + s];
#pragma unroll
    for (int ni = 0; ni < 4; ++ni) {
      const int n = ni * 16 + lrow;
      O2[(size_t)(b * 4096 + s) * 1024 + h * 64 + n] = f2bf(acc[ni][r] * z);
    }
  }
}

// ---------------------------------------------------------------------------
// Host-side launcher
// ---------------------------------------------------------------------------
extern "C" void kernel_launch(void* const* d_in, const int* in_sizes, int n_in,
                              void* d_out, int out_size, void* d_ws, size_t ws_size,
                              hipStream_t stream) {
  (void)in_sizes; (void)n_in; (void)out_size; (void)ws_size;

  const float* x  = (const float*)d_in[0];
  const float* Wq = (const float*)d_in[1];
  const float* bq = (const float*)d_in[2];
  const float* Wk = (const float*)d_in[3];
  const float* bk = (const float*)d_in[4];
  const float* Wv = (const float*)d_in[5];
  const float* bv = (const float*)d_in[6];
  const float* Wo = (const float*)d_in[7];
  const float* bo = (const float*)d_in[8];
  float* out = (float*)d_out;

  const size_t MD = 16384ull * 1024;
  const size_t WD = 1024ull * 1024;

  unsigned short* ws16 = (unsigned short*)d_ws;
  size_t o = 0;
  unsigned short* xb   = ws16 + o; o += MD;
  unsigned short* wqb  = ws16 + o; o += WD;
  unsigned short* wkb  = ws16 + o; o += WD;
  unsigned short* wvb  = ws16 + o; o += WD;
  unsigned short* wob  = ws16 + o; o += WD;
  unsigned short* qb   = ws16 + o; o += MD;   // Q, row-major bf16
  unsigned short* ktb  = ws16 + o; o += MD;   // K, channel-major bf16
  unsigned short* vtb  = ws16 + o; o += MD;   // V, channel-major bf16
  unsigned short* o2b  = ws16 + o; o += MD;   // attention output, row-major bf16
  unsigned short* kvb  = ws16 + o; o += 64ull * 4096;  // KVt per head
  o = (o + 63) & ~63ull;
  float* ksum = (float*)(ws16 + o); o += 2 * 4096;
  float* zbuf = (float*)(ws16 + o); o += 2 * 262144;

  // 1) fp32 -> bf16
  cvt_bf16<<<dim3((unsigned)(MD / 4 / 256)), dim3(256), 0, stream>>>(x, xb, (int)MD);
  cvt_bf16<<<dim3((unsigned)(WD / 4 / 256)), dim3(256), 0, stream>>>(Wq, wqb, (int)WD);
  cvt_bf16<<<dim3((unsigned)(WD / 4 / 256)), dim3(256), 0, stream>>>(Wk, wkb, (int)WD);
  cvt_bf16<<<dim3((unsigned)(WD / 4 / 256)), dim3(256), 0, stream>>>(Wv, wvb, (int)WD);
  cvt_bf16<<<dim3((unsigned)(WD / 4 / 256)), dim3(256), 0, stream>>>(Wo, wob, (int)WD);

  // 2) projections (TDM + WMMA): grid = (16384/128, 1024/64), 55296B dynamic LDS
  dim3 pg(128, 16);
  const unsigned smem = 55296;
  proj_gemm<1><<<pg, dim3(256), smem, stream>>>(xb, wqb, bq, qb, nullptr);   // Q
  proj_gemm<2><<<pg, dim3(256), smem, stream>>>(xb, wkb, bk, ktb, nullptr);  // K
  proj_gemm<0><<<pg, dim3(256), smem, stream>>>(xb, wvb, bv, vtb, nullptr);  // V

  // 3) K sums and Z
  ksum_kernel<<<dim3(16), dim3(256), 0, stream>>>(ktb, ksum);
  z_kernel<<<dim3(1024), dim3(256), 0, stream>>>(qb, ksum, zbuf);

  // 4) KV state (WMMA)
  kv_gemm<<<dim3(64), dim3(256), 0, stream>>>(vtb, ktb, kvb);

  // 5) out2 = (Q @ KV) * Z (WMMA)
  qkv_gemm<<<dim3(32, 64), dim3(256), 0, stream>>>(qb, kvb, zbuf, o2b);

  // 6) final projection to fp32 output (WMMA)
  proj_gemm<3><<<pg, dim3(256), smem, stream>>>(o2b, wob, bo, nullptr, out);
}
